// KitchenQuantizedLinear_84516366450871
// MI455X (gfx1250) — 8-bit matrix operand order repaired and checked by reading, compiled, not run on hardware
//
#include <hip/hip_runtime.h>
#include <hip/hip_bf16.h>

typedef int   v16i __attribute__((ext_vector_type(16)));
typedef float v8f  __attribute__((ext_vector_type(8)));

#define E4M3_MAX 448.0f

#if __has_builtin(__builtin_amdgcn_cvt_pk_fp8_f32)
#define HAVE_CVT_PK_FP8 1
#endif

__device__ __forceinline__ unsigned int f32_to_e4m3_sw(float x) {
    unsigned int u = __float_as_uint(x);
    unsigned int sign = (u >> 24) & 0x80u;
    unsigned int a = u & 0x7FFFFFFFu;
    if (a >= 0x7F800000u) return sign | 0x7Eu;
    float af = __uint_as_float(a);
    if (af >= 448.0f) return sign | 0x7Eu;
    if (af < 0.015625f) {
        int m = (int)rintf(af * 512.0f);
        return sign | (unsigned int)m;
    }
    unsigned int lsb = (a >> 20) & 1u;
    a += 0x7FFFFu + lsb;
    int e = (int)(a >> 23) - 127;
    unsigned int m3 = (a >> 20) & 7u;
    if (e > 8 || (e == 8 && m3 > 6u)) return sign | 0x7Eu;
    return sign | (unsigned int)((e + 7) << 3) | m3;
}

__device__ __forceinline__ unsigned int pk4_e4m3(float a, float b, float c, float d) {
#ifdef HAVE_CVT_PK_FP8
    int w = 0;
    w = __builtin_amdgcn_cvt_pk_fp8_f32(a, b, w, false);
    w = __builtin_amdgcn_cvt_pk_fp8_f32(c, d, w, true);
    return (unsigned int)w;
#else
    return  f32_to_e4m3_sw(a)        | (f32_to_e4m3_sw(b) << 8) |
           (f32_to_e4m3_sw(c) << 16) | (f32_to_e4m3_sw(d) << 24);
#endif
}

__global__ __launch_bounds__(256)
void amax_kernel(const float* __restrict__ p, size_t n4, unsigned int* __restrict__ out) {
    __shared__ float red[256];
    const float4* v = (const float4*)p;
    float m = 0.0f;
    for (size_t i = (size_t)blockIdx.x * 256 + threadIdx.x; i < n4;
         i += (size_t)gridDim.x * 256) {
        float4 f = v[i];
        m = fmaxf(m, fmaxf(fmaxf(fabsf(f.x), fabsf(f.y)),
                           fmaxf(fabsf(f.z), fabsf(f.w))));
    }
    red[threadIdx.x] = m;
    __syncthreads();
    for (int s = 128; s > 0; s >>= 1) {
        if ((int)threadIdx.x < s)
            red[threadIdx.x] = fmaxf(red[threadIdx.x], red[threadIdx.x + s]);
        __syncthreads();
    }
    if (threadIdx.x == 0) atomicMax(out, __float_as_uint(red[0]));
}

__global__ __launch_bounds__(256)
void quant_pack_a(const float* __restrict__ x, const float* __restrict__ amax,
                  unsigned char* __restrict__ apack, int M, int K) {
    size_t t = (size_t)blockIdx.x * 256 + threadIdx.x;
    size_t total = (size_t)M * K / 8;
    if (t >= total) return;
    float inv = E4M3_MAX / fmaxf(amax[0], 1e-12f);

    int    chunk  = (int)(t & 7);
    int    lane   = (int)((t >> 3) & 31);
    size_t tileId = t >> 8;
    int    nkt    = K >> 7;
    int    mt     = (int)(tileId / nkt);
    int    kt     = (int)(tileId % nkt);
    int    row    = mt * 16 + (lane & 15);
    int    kstart = kt * 128 + (chunk >> 1) * 32 + (lane >> 4) * 16 + (chunk & 1) * 8;

    const float4* src = (const float4*)(x + (size_t)row * K + kstart);
    float4 f0 = src[0], f1 = src[1];
    uint2 w;
    w.x = pk4_e4m3(f0.x * inv, f0.y * inv, f0.z * inv, f0.w * inv);
    w.y = pk4_e4m3(f1.x * inv, f1.y * inv, f1.z * inv, f1.w * inv);
    ((uint2*)apack)[t] = w;
}

__global__ __launch_bounds__(256)
void quant_pack_b(const float* __restrict__ w, const float* __restrict__ amax,
                  unsigned char* __restrict__ bpack, int N, int K) {
    size_t t = (size_t)blockIdx.x * 256 + threadIdx.x;
    size_t total = (size_t)N * K / 16;
    if (t >= total) return;
    float inv = E4M3_MAX / fmaxf(amax[0], 1e-12f);

    int    chunk  = (int)(t & 3);
    int    lane   = (int)((t >> 2) & 31);
    size_t tileId = t >> 7;
    int    nkt    = K >> 7;
    int    nt     = (int)(tileId / nkt);
    int    kt     = (int)(tileId % nkt);
    int    row    = nt * 16 + (lane & 15);
    int    kstart = kt * 128 + chunk * 32 + (lane >> 4) * 16;

    const float4* src = (const float4*)(w + (size_t)row * K + kstart);
    float4 f0 = src[0], f1 = src[1], f2 = src[2], f3 = src[3];
    uint4 o;
    o.x = pk4_e4m3(f0.x * inv, f0.y * inv, f0.z * inv, f0.w * inv);
    o.y = pk4_e4m3(f1.x * inv, f1.y * inv, f1.z * inv, f1.w * inv);
    o.z = pk4_e4m3(f2.x * inv, f2.y * inv, f2.z * inv, f2.w * inv);
    o.w = pk4_e4m3(f3.x * inv, f3.y * inv, f3.z * inv, f3.w * inv);
    ((uint4*)bpack)[t] = o;
}

union FragLoad { v16i v; int4 q[4]; };

__device__ __forceinline__ v16i load_frag(const unsigned char* base) {
    FragLoad f;
    const int4* p = (const int4*)base;
    f.q[0] = p[0]; f.q[1] = p[1]; f.q[2] = p[2]; f.q[3] = p[3];
    return f.v;
}

__global__ __launch_bounds__(256)
void fp8_gemm_kernel(const unsigned char* __restrict__ apack,
                     const unsigned char* __restrict__ bpack,
                     const float* __restrict__ bias,
                     const float* __restrict__ amax2,
                     float* __restrict__ out,
                     int M, int N, int K) {
    const int nkt  = K >> 7;
    const int lane = threadIdx.x & 31;
    const int wave = threadIdx.x >> 5;
    const int wm   = wave >> 2;
    const int wn   = wave & 3;

    const int mt0 = blockIdx.y * 8 + wm * 4;
    const int nt0 = blockIdx.x * 8 + wn * 2;

    const unsigned char* aBase[4];
    const unsigned char* bBase[2];
#pragma unroll
    for (int i = 0; i < 4; ++i)
        aBase[i] = apack + ((size_t)(mt0 + i) * nkt) * 2048 + (size_t)lane * 64;
#pragma unroll
    for (int j = 0; j < 2; ++j)
        bBase[j] = bpack + ((size_t)(nt0 + j) * nkt) * 2048 + (size_t)lane * 64;

    v8f acc[4][2];
#pragma unroll
    for (int i = 0; i < 4; ++i)
#pragma unroll
        for (int j = 0; j < 2; ++j) acc[i][j] = (v8f)(0.0f);

    for (int kt = 0; kt < nkt; ++kt) {
        const size_t koff = (size_t)kt * 2048;
        v16i af[4], bf[2];
#pragma unroll
        for (int i = 0; i < 4; ++i) af[i] = load_frag(aBase[i] + koff);
#pragma unroll
        for (int j = 0; j < 2; ++j) bf[j] = load_frag(bBase[j] + koff);
#pragma unroll
        for (int i = 0; i < 4; ++i)
#pragma unroll
            for (int j = 0; j < 2; ++j)
                acc[i][j] = __builtin_amdgcn_wmma_f32_16x16x128_fp8_fp8(
                    af[i], bf[j], (short)0, acc[i][j], false, false);
    }

    const float sx = fmaxf(amax2[0], 1e-12f) / E4M3_MAX;
    const float sw = fmaxf(amax2[1], 1e-12f) / E4M3_MAX;
    const float s  = sx * sw;

    const int colInTile = lane & 15;
    const int rowOff    = (lane >> 4) * 8;
#pragma unroll
    for (int j = 0; j < 2; ++j) {
        const int col = (nt0 + j) * 16 + colInTile;
        const float bv = bias[col];
#pragma unroll
        for (int i = 0; i < 4; ++i) {
            const int rowBase = (mt0 + i) * 16 + rowOff;
#pragma unroll
            for (int r = 0; r < 8; ++r)
                out[(size_t)(rowBase + r) * N + col] = acc[i][j][r] * s + bv;
        }
    }
}

extern "C" void kernel_launch(void* const* d_in, const int* in_sizes, int n_in,
                              void* d_out, int out_size, void* d_ws, size_t ws_size,
                              hipStream_t stream) {
    const float* x = (const float*)d_in[0];
    const float* W = (const float*)d_in[1];
    const float* b = (const float*)d_in[2];
    float* out = (float*)d_out;

    const int N = in_sizes[2];
    const int K = in_sizes[1] / N;
    const int M = in_sizes[0] / K;

    unsigned char* ws    = (unsigned char*)d_ws;
    unsigned int*  amax  = (unsigned int*)ws;
    unsigned char* apack = ws + 256;
    unsigned char* bpack = apack + (size_t)M * K;

    hipMemsetAsync(amax, 0, 2 * sizeof(unsigned int), stream);

    amax_kernel<<<1024, 256, 0, stream>>>(x, (size_t)M * K / 4, amax + 0);
    amax_kernel<<<512,  256, 0, stream>>>(W, (size_t)N * K / 4, amax + 1);

    {
        size_t tot = (size_t)M * K / 8;
        quant_pack_a<<<(unsigned)((tot + 255) / 256), 256, 0, stream>>>(
            x, (const float*)amax, apack, M, K);
    }
    {
        size_t tot = (size_t)N * K / 16;
        quant_pack_b<<<(unsigned)((tot + 255) / 256), 256, 0, stream>>>(
            W, (const float*)(amax + 1), bpack, N, K);
    }

    dim3 grid(N / 128, M / 128, 1);
    fp8_gemm_kernel<<<grid, 256, 0, stream>>>(
        apack, bpack, b, (const float*)amax, out, M, N, K);
}
